// GAT_86139864089310
// MI455X (gfx1250) — matrix operand order repaired and checked by reading, not compiled, not run on hardware
//
#include <hip/hip_runtime.h>
#include <math.h>

// ---------------- CDNA5 WMMA / TDM types ----------------
typedef __attribute__((ext_vector_type(16))) __bf16 v16bf;
typedef __attribute__((ext_vector_type(8)))  float  v8f;
typedef __attribute__((ext_vector_type(4)))  unsigned int u32x4;
typedef __attribute__((ext_vector_type(8)))  int    i32x8;
typedef __attribute__((ext_vector_type(4)))  int    i32x4;

#define FIN   128
#define C1TOT 32      // H1*C1 = 2*16
#define C2TOT 40      // H2*C2 = 1*40
#define ROWS_PER_BLK 128   // 8 waves x 16 rows

static __device__ __forceinline__ __bf16 f2bf(float f) {
    union { float f; unsigned u; } v; v.f = f;
    unsigned r = v.u + 0x7FFFu + ((v.u >> 16) & 1u);   // round-to-nearest-even
    unsigned short h = (unsigned short)(r >> 16);
    __bf16 o; __builtin_memcpy(&o, &h, 2); return o;
}

static __device__ __forceinline__ void atomicMaxFloat(float* addr, float val) {
    // sign-magnitude trick: max on int for >=0, min on uint for <0
    if (val >= 0.0f) atomicMax((int*)addr, __float_as_int(val));
    else             atomicMin((unsigned int*)addr, (unsigned int)__float_as_int(val));
}

static __device__ __forceinline__ void pack8(v16bf& a, int base, float4 lo, float4 hi4) {
    a[base + 0] = f2bf(lo.x);  a[base + 1] = f2bf(lo.y);
    a[base + 2] = f2bf(lo.z);  a[base + 3] = f2bf(lo.w);
    a[base + 4] = f2bf(hi4.x); a[base + 5] = f2bf(hi4.y);
    a[base + 6] = f2bf(hi4.z); a[base + 7] = f2bf(hi4.w);
}

// ---------------- init kernels ----------------
__global__ void gat_init_nodes(float* __restrict__ agg1, const float* __restrict__ b1,
                               float* __restrict__ m1, float* __restrict__ s1,
                               float* __restrict__ m2, float* __restrict__ s2, int n) {
    int i = blockIdx.x * blockDim.x + threadIdx.x;
    if (i < n * C1TOT) agg1[i] = b1[i & (C1TOT - 1)];     // out = sum + bias
    if (i < n * 2)   { m1[i] = __int_as_float(0xff800000); s1[i] = 0.0f; }
    if (i < n)       { m2[i] = __int_as_float(0xff800000); s2[i] = 0.0f; }
}

__global__ void gat_init_out(float* __restrict__ out, const float* __restrict__ b2, int n) {
    int i = blockIdx.x * blockDim.x + threadIdx.x;
    if (i < n * C2TOT) out[i] = b2[i % C2TOT];
}

// ---------------- GEMM1: h1[N,32] = X[N,128] @ W1[128,32]  (TDM + bf16 WMMA) ----------------
// A panel (128 rows x 128 f32 = 64KB) staged into LDS by the Tensor Data Mover;
// B fragments pre-swizzled in LDS to the exact WMMA B-layout (2 x ds_load_b128 each).
__global__ void gat_gemm1_wmma(const float* __restrict__ X, const float* __restrict__ W,
                               float* __restrict__ H1out, int n) {
    __shared__ __attribute__((aligned(64))) float  Ards[ROWS_PER_BLK * FIN];  // 64 KB
    __shared__ __attribute__((aligned(64))) __bf16 Wlds[8 * 32 * 16];         //  8 KB

    const int t    = threadIdx.x;
    const int lane = t & 31;
    const int m    = lane & 15;
    const int hi   = lane >> 4;
    const int rowBlock0 = blockIdx.x * ROWS_PER_BLK;

#if __has_builtin(__builtin_amdgcn_tensor_load_to_lds)
    if (t < 32) {                                   // wave 0 launches the TDM
        unsigned long long ga = (unsigned long long)(const void*)(X + (size_t)rowBlock0 * FIN);
        unsigned int lds32 = (unsigned int)(size_t)(void*)&Ards[0];
        unsigned int rowsRem = (unsigned int)(n - rowBlock0);   // OOB rows read as zero

        u32x4 g0;
        g0[0] = 1u;                                             // count=1, no gather
        g0[1] = lds32;                                          // lds_addr
        g0[2] = (unsigned int)ga;                               // global_addr[31:0]
        g0[3] = ((unsigned int)(ga >> 32) & 0x01FFFFFFu) | (2u << 30); // addr[56:32] | type=2

        i32x8 g1;
        g1[0] = (int)(2u << 16);                                // data_size = 4B
        g1[1] = (int)((unsigned)FIN << 16);                     // tensor_dim0 lo16
        g1[2] = (int)((rowsRem & 0xFFFFu) << 16);               // dim0 hi16=0 | tensor_dim1 lo16
        g1[3] = (int)(((rowsRem >> 16) & 0xFFFFu) | ((unsigned)FIN << 16)); // dim1 hi | tile_dim0
        g1[4] = ROWS_PER_BLK;                                   // tile_dim1 | tile_dim2=0
        g1[5] = FIN;                                            // tensor_dim0_stride lo32
        g1[6] = 0;                                              // stride0 hi | stride1 lo
        g1[7] = 0;

        i32x4 g2 = {0, 0, 0, 0};                                // 2D: groups 2/3 zero
        i32x4 g3 = {0, 0, 0, 0};
#if __clang_major__ >= 23
        i32x8 g4 = {0, 0, 0, 0, 0, 0, 0, 0};
        __builtin_amdgcn_tensor_load_to_lds(g0, g1, g2, g3, g4, 0);
#else
        __builtin_amdgcn_tensor_load_to_lds(g0, g1, g2, g3, 0);
#endif
    }
#else
    // fallback: cooperative staging
    for (int idx = t; idx < ROWS_PER_BLK * FIN; idx += 256) {
        int r = rowBlock0 + (idx >> 7);
        Ards[idx] = (r < n) ? X[(size_t)r * FIN + (idx & (FIN - 1))] : 0.0f;
    }
#endif

    // convert/swizzle W while the DMA is in flight:
    // 256 threads x 16 halfs = 8 frags * 32 lanes * 16
    {
        int f = t >> 5, L = t & 31;
        int kk = f >> 1, ct = f & 1;
        int lhi = L >> 4, lm = L & 15;
        v16bf w;
#pragma unroll
        for (int j = 0; j < 16; ++j) {
            int k = kk * 32 + lhi * 8 + (j < 8 ? j : 8 + j);   // B half j <-> K = hi*8 + j (j<8), 16 + hi*8 + (j-8) (j>=8)
            w[j] = f2bf(W[k * C1TOT + ct * 16 + lm]);
        }
        *reinterpret_cast<v16bf*>(&Wlds[(f * 32 + L) * 16]) = w;
    }

#if __has_builtin(__builtin_amdgcn_tensor_load_to_lds)
    if (t < 32) __builtin_amdgcn_s_wait_tensorcnt(0);
#endif
    __syncthreads();

    const int row0 = rowBlock0 + (t >> 5) * 16;           // this wave's 16-row tile
    if (row0 >= n) return;
    const float* xr = Ards + ((t >> 5) * 16 + m) * FIN;   // LDS-resident A row

    v8f acc0 = {}; v8f acc1 = {};
#pragma unroll
    for (int kk = 0; kk < 4; ++kk) {
        const int k0 = kk * 32;
        const float4* p0 = (const float4*)(xr + k0 + hi * 8);
        const float4* p1 = (const float4*)(xr + k0 + 16 + hi * 8);
        v16bf a;
        pack8(a, 0, p0[0], p0[1]);
        pack8(a, 8, p1[0], p1[1]);

        v16bf b0 = *reinterpret_cast<const v16bf*>(&Wlds[((kk * 2 + 0) * 32 + lane) * 16]);
        v16bf b1 = *reinterpret_cast<const v16bf*>(&Wlds[((kk * 2 + 1) * 32 + lane) * 16]);
        acc0 = __builtin_amdgcn_wmma_f32_16x16x32_bf16(false, a, false, b0, (short)0, acc0, false, false);
        acc1 = __builtin_amdgcn_wmma_f32_16x16x32_bf16(false, a, false, b1, (short)0, acc1, false, false);
    }
    // D layout: VGPR i -> M = i + 8*hi, N = m
#pragma unroll
    for (int i = 0; i < 8; ++i) {
        int r = row0 + i + hi * 8;
        if (r < n) {
            H1out[(size_t)r * C1TOT +  0 + m] = acc0[i];
            H1out[(size_t)r * C1TOT + 16 + m] = acc1[i];
        }
    }
}

// ---------------- GEMM2: h2[N,40] = X2[N,32] @ W2[32,40]  (bf16 WMMA) ----------------
__global__ void gat_gemm2_wmma(const float* __restrict__ X2, const float* __restrict__ W,
                               float* __restrict__ H2out, int n) {
    __shared__ __attribute__((aligned(64))) __bf16 Wlds[3 * 32 * 16];   // 3 KB

    const int t    = threadIdx.x;
    const int lane = t & 31;
    const int m    = lane & 15;
    const int hi   = lane >> 4;

    if (t < 96) {                                        // 3 frags * 32 lanes
        int ct = t >> 5, L = t & 31;
        int lhi = L >> 4, lm = L & 15;
        int col = ct * 16 + lm;
        v16bf w;
#pragma unroll
        for (int j = 0; j < 16; ++j) {
            int k = lhi * 8 + (j < 8 ? j : 8 + j);
            w[j] = (col < C2TOT) ? f2bf(W[k * C2TOT + col]) : f2bf(0.0f);
        }
        *reinterpret_cast<v16bf*>(&Wlds[(ct * 32 + L) * 16]) = w;
    }
    __syncthreads();

    const int wave = (blockIdx.x * blockDim.x + t) >> 5;
    const int row0 = wave * 16;
    if (row0 >= n) return;

    const int row = min(row0 + m, n - 1);
    const float* xr = X2 + (size_t)row * C1TOT;

    const float4* p0 = (const float4*)(xr + hi * 8);
    const float4* p1 = (const float4*)(xr + 16 + hi * 8);
    v16bf a;
    pack8(a, 0, p0[0], p0[1]);
    pack8(a, 8, p1[0], p1[1]);

    v8f acc[3] = {{}, {}, {}};
#pragma unroll
    for (int ct = 0; ct < 3; ++ct) {
        v16bf b = *reinterpret_cast<const v16bf*>(&Wlds[(ct * 32 + lane) * 16]);
        acc[ct] = __builtin_amdgcn_wmma_f32_16x16x32_bf16(false, a, false, b, (short)0, acc[ct], false, false);
    }
#pragma unroll
    for (int ct = 0; ct < 3; ++ct) {
        int col = ct * 16 + m;
        if (col < C2TOT) {
#pragma unroll
            for (int i = 0; i < 8; ++i) {
                int r = row0 + i + hi * 8;
                if (r < n) H2out[(size_t)r * C2TOT + col] = acc[ct][i];
            }
        }
    }
}

// ---------------- per-node attention scalars ----------------
__global__ void gat_att1(const float* __restrict__ H1buf, const float* __restrict__ att_src,
                         const float* __restrict__ att_dst, float* __restrict__ asrc,
                         float* __restrict__ adst, int n) {
    int i = blockIdx.x * blockDim.x + threadIdx.x;     // over n*2 (heads)
    if (i >= n * 2) return;
    int node = i >> 1, h = i & 1;
    const float4* hp = (const float4*)(H1buf + (size_t)node * C1TOT + h * 16);
    const float4* as = (const float4*)(att_src + h * 16);
    const float4* ad = (const float4*)(att_dst + h * 16);
    float s = 0.0f, d = 0.0f;
#pragma unroll
    for (int c = 0; c < 4; ++c) {
        float4 hv = hp[c], av = as[c], dv = ad[c];
        s += hv.x * av.x + hv.y * av.y + hv.z * av.z + hv.w * av.w;
        d += hv.x * dv.x + hv.y * dv.y + hv.z * dv.z + hv.w * dv.w;
    }
    asrc[i] = s; adst[i] = d;
}

__global__ void gat_att2(const float* __restrict__ H2buf, const float* __restrict__ att_src,
                         const float* __restrict__ att_dst, float* __restrict__ asrc,
                         float* __restrict__ adst, int n) {
    int i = blockIdx.x * blockDim.x + threadIdx.x;
    if (i >= n) return;
    const float* hp = H2buf + (size_t)i * C2TOT;
    float s = 0.0f, d = 0.0f;
#pragma unroll
    for (int c = 0; c < C2TOT; ++c) { s += hp[c] * att_src[c]; d += hp[c] * att_dst[c]; }
    asrc[i] = s; adst[i] = d;
}

// ---------------- edge helpers ----------------
static __device__ __forceinline__ void edge_ids(const int* __restrict__ srcI,
                                                const int* __restrict__ dstI,
                                                int i, int e_real, int& s, int& d) {
    if (i < e_real) { s = srcI[i]; d = dstI[i]; }
    else            { s = i - e_real; d = s; }        // self-loops appended
}

// ---------------- edge passes, layer 1 (H=2) ----------------
__global__ void gat_edge_max1(const int* __restrict__ srcI, const int* __restrict__ dstI,
                              const float* __restrict__ asrc, const float* __restrict__ adst,
                              float* __restrict__ mbuf, int etot, int e_real) {
    int i = blockIdx.x * blockDim.x + threadIdx.x;
    if (i >= etot) return;
    if (i + 4096 < e_real) __builtin_prefetch(&srcI[i + 4096], 0, 1);
    int s, d; edge_ids(srcI, dstI, i, e_real, s, d);
#pragma unroll
    for (int h = 0; h < 2; ++h) {
        float e = asrc[s * 2 + h] + adst[d * 2 + h];
        e = e > 0.0f ? e : 0.2f * e;
        atomicMaxFloat(&mbuf[d * 2 + h], e);
    }
}

__global__ void gat_edge_sum1(const int* __restrict__ srcI, const int* __restrict__ dstI,
                              const float* __restrict__ asrc, const float* __restrict__ adst,
                              const float* __restrict__ mbuf, float* __restrict__ sbuf,
                              int etot, int e_real) {
    int i = blockIdx.x * blockDim.x + threadIdx.x;
    if (i >= etot) return;
    int s, d; edge_ids(srcI, dstI, i, e_real, s, d);
#pragma unroll
    for (int h = 0; h < 2; ++h) {
        float e = asrc[s * 2 + h] + adst[d * 2 + h];
        e = e > 0.0f ? e : 0.2f * e;
        atomicAdd(&sbuf[d * 2 + h], __expf(e - mbuf[d * 2 + h]));
    }
}

__global__ void gat_edge_agg1(const int* __restrict__ srcI, const int* __restrict__ dstI,
                              const float* __restrict__ asrc, const float* __restrict__ adst,
                              const float* __restrict__ mbuf, const float* __restrict__ sbuf,
                              const float* __restrict__ H1buf, float* __restrict__ agg,
                              int etot, int e_real) {
    int i = blockIdx.x * blockDim.x + threadIdx.x;
    if (i >= etot) return;
    if (i + 4096 < e_real) __builtin_prefetch(&srcI[i + 4096], 0, 1);
    int s, d; edge_ids(srcI, dstI, i, e_real, s, d);
    const float4* hs = (const float4*)(H1buf + (size_t)s * C1TOT);
    float* ag = agg + (size_t)d * C1TOT;
#pragma unroll
    for (int h = 0; h < 2; ++h) {
        float e = asrc[s * 2 + h] + adst[d * 2 + h];
        e = e > 0.0f ? e : 0.2f * e;
        float alpha = __expf(e - mbuf[d * 2 + h]) / (sbuf[d * 2 + h] + 1e-16f);
#pragma unroll
        for (int q = 0; q < 4; ++q) {
            float4 hv = hs[h * 4 + q];
            int c = h * 16 + q * 4;
            atomicAdd(&ag[c + 0], hv.x * alpha);
            atomicAdd(&ag[c + 1], hv.y * alpha);
            atomicAdd(&ag[c + 2], hv.z * alpha);
            atomicAdd(&ag[c + 3], hv.w * alpha);
        }
    }
}

__global__ void gat_relu(float* __restrict__ x, int n) {
    int i = blockIdx.x * blockDim.x + threadIdx.x;
    if (i < n) x[i] = fmaxf(x[i], 0.0f);
}

// ---------------- edge passes, layer 2 (H=1, C=40) ----------------
__global__ void gat_edge_max2(const int* __restrict__ srcI, const int* __restrict__ dstI,
                              const float* __restrict__ asrc, const float* __restrict__ adst,
                              float* __restrict__ mbuf, int etot, int e_real) {
    int i = blockIdx.x * blockDim.x + threadIdx.x;
    if (i >= etot) return;
    int s, d; edge_ids(srcI, dstI, i, e_real, s, d);
    float e = asrc[s] + adst[d];
    e = e > 0.0f ? e : 0.2f * e;
    atomicMaxFloat(&mbuf[d], e);
}

__global__ void gat_edge_sum2(const int* __restrict__ srcI, const int* __restrict__ dstI,
                              const float* __restrict__ asrc, const float* __restrict__ adst,
                              const float* __restrict__ mbuf, float* __restrict__ sbuf,
                              int etot, int e_real) {
    int i = blockIdx.x * blockDim.x + threadIdx.x;
    if (i >= etot) return;
    int s, d; edge_ids(srcI, dstI, i, e_real, s, d);
    float e = asrc[s] + adst[d];
    e = e > 0.0f ? e : 0.2f * e;
    atomicAdd(&sbuf[d], __expf(e - mbuf[d]));
}

__global__ void gat_edge_agg2(const int* __restrict__ srcI, const int* __restrict__ dstI,
                              const float* __restrict__ asrc, const float* __restrict__ adst,
                              const float* __restrict__ mbuf, const float* __restrict__ sbuf,
                              const float* __restrict__ H2buf, float* __restrict__ out,
                              int etot, int e_real) {
    int i = blockIdx.x * blockDim.x + threadIdx.x;
    if (i >= etot) return;
    if (i + 4096 < e_real) __builtin_prefetch(&srcI[i + 4096], 0, 1);
    int s, d; edge_ids(srcI, dstI, i, e_real, s, d);
    float e = asrc[s] + adst[d];
    e = e > 0.0f ? e : 0.2f * e;
    float alpha = __expf(e - mbuf[d]) / (sbuf[d] + 1e-16f);
    const float4* hs = (const float4*)(H2buf + (size_t)s * C2TOT);
    float* op = out + (size_t)d * C2TOT;
#pragma unroll
    for (int q = 0; q < 10; ++q) {
        float4 hv = hs[q];
        atomicAdd(&op[q * 4 + 0], hv.x * alpha);
        atomicAdd(&op[q * 4 + 1], hv.y * alpha);
        atomicAdd(&op[q * 4 + 2], hv.z * alpha);
        atomicAdd(&op[q * 4 + 3], hv.w * alpha);
    }
}

// ---------------- launcher ----------------
extern "C" void kernel_launch(void* const* d_in, const int* in_sizes, int n_in,
                              void* d_out, int out_size, void* d_ws, size_t ws_size,
                              hipStream_t stream) {
    const float* features = (const float*)d_in[0];
    const int*   edges    = (const int*)d_in[1];
    const float* W1       = (const float*)d_in[3];
    const float* att_src1 = (const float*)d_in[4];
    const float* att_dst1 = (const float*)d_in[5];
    const float* b1       = (const float*)d_in[6];
    const float* W2       = (const float*)d_in[7];
    const float* att_src2 = (const float*)d_in[8];
    const float* att_dst2 = (const float*)d_in[9];
    const float* b2       = (const float*)d_in[10];

    const int N = in_sizes[0] / FIN;
    const int E = in_sizes[1] / 2;
    const int* srcIdx = edges;
    const int* dstIdx = edges + E;
    const int etot = E + N;

    // workspace carve-out (floats): h1 | agg1/x2 | h2 | asrc1 adst1 m1 s1 | asrc2 adst2 m2 s2
    float* h1    = (float*)d_ws;
    float* agg1  = h1    + (size_t)N * C1TOT;
    float* h2    = agg1  + (size_t)N * C1TOT;
    float* asrc1 = h2    + (size_t)N * C2TOT;
    float* adst1 = asrc1 + (size_t)N * 2;
    float* m1    = adst1 + (size_t)N * 2;
    float* s1    = m1    + (size_t)N * 2;
    float* asrc2 = s1    + (size_t)N * 2;
    float* adst2 = asrc2 + (size_t)N;
    float* m2    = adst2 + (size_t)N;
    float* s2    = m2    + (size_t)N;
    float* out   = (float*)d_out;

    const int B = 256;
    auto cdiv = [](int a, int b) { return (a + b - 1) / b; };
    const int gemm1Blocks = cdiv(N, ROWS_PER_BLK);        // 128 rows / block (TDM tile)
    const int gemm2Blocks = cdiv(cdiv(N, 16) * 32, B);    // 1 wave per 16-row tile

    gat_init_nodes<<<cdiv(N * C1TOT, B), B, 0, stream>>>(agg1, b1, m1, s1, m2, s2, N);
    gat_init_out  <<<cdiv(N * C2TOT, B), B, 0, stream>>>(out, b2, N);

    // ---- layer 1 ----
    gat_gemm1_wmma<<<gemm1Blocks, B, 0, stream>>>(features, W1, h1, N);
    gat_att1      <<<cdiv(N * 2, B), B, 0, stream>>>(h1, att_src1, att_dst1, asrc1, adst1, N);
    gat_edge_max1 <<<cdiv(etot, B), B, 0, stream>>>(srcIdx, dstIdx, asrc1, adst1, m1, etot, E);
    gat_edge_sum1 <<<cdiv(etot, B), B, 0, stream>>>(srcIdx, dstIdx, asrc1, adst1, m1, s1, etot, E);
    gat_edge_agg1 <<<cdiv(etot, B), B, 0, stream>>>(srcIdx, dstIdx, asrc1, adst1, m1, s1, h1, agg1, etot, E);
    gat_relu      <<<cdiv(N * C1TOT, B), B, 0, stream>>>(agg1, N * C1TOT);

    // ---- layer 2 ----
    gat_gemm2_wmma<<<gemm2Blocks, B, 0, stream>>>(agg1, W2, h2, N);
    gat_att2      <<<cdiv(N, B), B, 0, stream>>>(h2, att_src2, att_dst2, asrc2, adst2, N);
    gat_edge_max2 <<<cdiv(etot, B), B, 0, stream>>>(srcIdx, dstIdx, asrc2, adst2, m2, etot, E);
    gat_edge_sum2 <<<cdiv(etot, B), B, 0, stream>>>(srcIdx, dstIdx, asrc2, adst2, m2, s2, etot, E);
    gat_edge_agg2 <<<cdiv(etot, B), B, 0, stream>>>(srcIdx, dstIdx, asrc2, adst2, m2, s2, h2, out, etot, E);
}
